// LIFCGBenchmark_944892805202
// MI455X (gfx1250) — matrix operand order repaired and checked by reading, not compiled, not run on hardware
//
#include <hip/hip_runtime.h>
#include <stdint.h>

typedef __attribute__((ext_vector_type(16))) __bf16 v16bf;
typedef __attribute__((ext_vector_type(8)))  __bf16 v8bf;
typedef __attribute__((ext_vector_type(8)))  float  v8f;

#define T_STEPS 1024
#define BATCH   256
#define FDIM    512
#define ROW_E   520   // padded LDS row stride in bf16 elements (1040 B -> bank spread)

union V16U { v16bf v; v8bf h[2]; };

__device__ __forceinline__ unsigned short f2bf_rne(float f) {
  unsigned u = __float_as_uint(f);
  u += 0x7FFFu + ((u >> 16) & 1u);
  return (unsigned short)(u >> 16);
}

__global__ __launch_bounds__(128) void lif_fused_wmma_kernel(
    const float* __restrict__ spikes,   // [T,B,F] f32, values in {0,1}
    const float* __restrict__ W,        // [F,F] f32 row-major (g,f)
    float* __restrict__ out)            // [3,B,F]: z_last, v, i
{
  __shared__ unsigned short sA[2][16 * ROW_E];  // double-buffered spike tile, bf16
  __shared__ unsigned short sW[128 * ROW_E];    // W slice (128 cols x 512 K), bf16

  const int tid  = threadIdx.x;
  const int lane = tid & 31;
  const int wid  = tid >> 5;     // 0..3 (wave id)
  const int hh   = lane >> 4;    // lane half (ISA fragment layout)
  const int ln   = lane & 15;

  const int bbase  = blockIdx.x * 16;    // batch rows for this WG
  const int gblock = blockIdx.y * 128;   // output columns for this WG

  // ---- one-time: stage this WG's W slice into LDS as bf16 (RNE), col-major rows
  {
    const int c = tid;  // one column (output neuron g) per thread
    const float4* wsrc = (const float4*)(W + (size_t)(gblock + c) * FDIM);
    for (int l = 0; l < 128; ++l) {
      float4 w = wsrc[l];
      uint2 p;
      p.x = (unsigned)f2bf_rne(w.x) | ((unsigned)f2bf_rne(w.y) << 16);
      p.y = (unsigned)f2bf_rne(w.z) | ((unsigned)f2bf_rne(w.w) << 16);
      *(uint2*)&sW[c * ROW_E + l * 4] = p;
    }
  }

  // The 16x512 f32 spike tile for one timestep is fully contiguous (32 KB).
  // Flat float4 index tid + l*128  ->  row m = l, col f = tid*4:
  //   * global loads: 32 consecutive float4 per wave instruction (512 B, coalesced)
  //   * LDS publish at [l*ROW_E + tid*4]: consecutive lanes -> consecutive 8 B
  //     (bank-conflict-free stores)
  const float4* tile0 = (const float4*)(spikes + (size_t)bbase * FDIM);
  float4 stage[16];
  {
    #pragma unroll
    for (int l = 0; l < 16; ++l) stage[l] = tile0[tid + l * 128];
  }

  // ---- LIF state resident in accumulator fragments (C/D layout)
  v8f vacc[2], iacc[2], zfr[2];
  #pragma unroll
  for (int j = 0; j < 2; ++j)
    #pragma unroll
    for (int r = 0; r < 8; ++r) { vacc[j][r] = 0.f; iacc[j][r] = 0.f; zfr[j][r] = 0.f; }

  // prologue: publish tile 0 into buffer 0 (f32->bf16 truncation exact for {0,1})
  {
    #pragma unroll
    for (int l = 0; l < 16; ++l) {
      float4 s = stage[l];
      uint2 p;
      p.x = (__float_as_uint(s.x) >> 16) | (__float_as_uint(s.y) & 0xFFFF0000u);
      p.y = (__float_as_uint(s.z) >> 16) | (__float_as_uint(s.w) & 0xFFFF0000u);
      *(uint2*)&sA[0][l * ROW_E + tid * 4] = p;
    }
  }
  __syncthreads();

  int p = 0;
  for (int t = 0; t < T_STEPS; ++t) {
    // (a) issue global loads for tile t+1 early; consumed only after the WMMA chain
    if (t + 1 < T_STEPS) {
      const float4* src = (const float4*)(spikes + (size_t)(t + 1) * BATCH * FDIM
                                          + (size_t)bbase * FDIM);
      #pragma unroll
      for (int l = 0; l < 16; ++l) stage[l] = src[tid + l * 128];
    }
    if (t + 2 < T_STEPS)
      __builtin_prefetch(spikes + (size_t)(t + 2) * BATCH * FDIM
                         + (size_t)bbase * FDIM + tid * 64, 0, 1);

    // (b) LIF: leak/integrate, threshold, reset, synaptic decay (pre-step state)
    #pragma unroll
    for (int j = 0; j < 2; ++j) {
      #pragma unroll
      for (int r = 0; r < 8; ++r) {
        float vdec = 0.9f * vacc[j][r] + 0.1f * iacc[j][r];  // v + dt*tau_m*((0-v)+i)
        float z = (vdec > 1.0f) ? 1.0f : 0.0f;               // heaviside(v - v_th)
        zfr[j][r]  = z;
        vacc[j][r] = vdec - z * vdec;                        // (1-z)*vdec, V_RESET=0
        iacc[j][r] = 0.8f * iacc[j][r];                      // i - dt*tau_s*i
      }
    }

    //     i += spikes_t @ W^T : K=512 as 16 chained bf16 WMMAs per N-tile
    const unsigned short* sAp = sA[p];
    #pragma unroll
    for (int kc = 0; kc < 16; ++kc) {
      V16U a, b0, b1;
      // A 16x32 bf16 layout: lane group hh = K half, 16 contiguous K per lane
      const unsigned short* pa = &sAp[ln * ROW_E + kc * 32 + 16 * hh];
      a.h[0] = *(const v8bf*)(pa);
      a.h[1] = *(const v8bf*)(pa + 8);
      // B 32x16 bf16 layout: lane group hh = K half, 16 contiguous K per lane
      const int c0 = wid * 32 + ln;
      const unsigned short* pb0 = &sW[c0 * ROW_E + kc * 32 + 16 * hh];
      b0.h[0] = *(const v8bf*)(pb0);
      b0.h[1] = *(const v8bf*)(pb0 + 8);
      const unsigned short* pb1 = pb0 + 16 * ROW_E;
      b1.h[0] = *(const v8bf*)(pb1);
      b1.h[1] = *(const v8bf*)(pb1 + 8);
      iacc[0] = __builtin_amdgcn_wmma_f32_16x16x32_bf16(false, a.v, false, b0.v,
                                                        (short)0, iacc[0], false, false);
      iacc[1] = __builtin_amdgcn_wmma_f32_16x16x32_bf16(false, a.v, false, b1.v,
                                                        (short)0, iacc[1], false, false);
    }

    // (c) publish tile t+1 into the other buffer (waits on (a) loads only here)
    if (t + 1 < T_STEPS) {
      unsigned short* dst = sA[p ^ 1];
      #pragma unroll
      for (int l = 0; l < 16; ++l) {
        float4 s = stage[l];
        uint2 q;
        q.x = (__float_as_uint(s.x) >> 16) | (__float_as_uint(s.y) & 0xFFFF0000u);
        q.y = (__float_as_uint(s.z) >> 16) | (__float_as_uint(s.w) & 0xFFFF0000u);
        *(uint2*)&dst[l * ROW_E + tid * 4] = q;
      }
    }
    // (d) single barrier per step: makes t+1 tile visible AND protects buffer reuse
    __syncthreads();
    p ^= 1;
  }

  // ---- write z_last, v, i (C/D layout: M = r + 8*hh, N = ln)
  const size_t BF = (size_t)BATCH * FDIM;
  #pragma unroll
  for (int j = 0; j < 2; ++j) {
    const int g = gblock + wid * 32 + j * 16 + ln;
    #pragma unroll
    for (int r = 0; r < 8; ++r) {
      const int b = bbase + r + 8 * hh;
      out[(size_t)b * FDIM + g]            = zfr[j][r];
      out[BF + (size_t)b * FDIM + g]       = vacc[j][r];
      out[2 * BF + (size_t)b * FDIM + g]   = iacc[j][r];
    }
  }
}

extern "C" void kernel_launch(void* const* d_in, const int* in_sizes, int n_in,
                              void* d_out, int out_size, void* d_ws, size_t ws_size,
                              hipStream_t stream) {
  (void)in_sizes; (void)n_in; (void)d_ws; (void)ws_size; (void)out_size;
  const float* spikes = (const float*)d_in[0];   // [1024,256,512] f32
  const float* W      = (const float*)d_in[1];   // [512,512] f32
  float* out          = (float*)d_out;           // [3,256,512] f32
  dim3 grid(BATCH / 16, FDIM / 128);             // 16 x 4 = 64 workgroups
  dim3 block(128);                               // 4 wave32 per WG
  hipLaunchKernelGGL(lif_fused_wmma_kernel, grid, block, 0, stream, spikes, W, out);
}
